// QBWBottleneck_28381143892470
// MI455X (gfx1250) — matrix operand order repaired and checked by reading, not compiled, not run on hardware
//
#include <hip/hip_runtime.h>

// ---- problem constants ----
#define NIMG   128
#define IN_C   512
#define MID_C  128
#define OUT_G  32
#define OUT_C  544          // 512 + 32
#define HW     1024         // 32*32
#define EPSV   1e-5f

typedef __attribute__((ext_vector_type(16))) __bf16 v16bf;
typedef __attribute__((ext_vector_type(8)))  __bf16 v8bf;
typedef __attribute__((ext_vector_type(8)))  float  v8f;
typedef int v4i __attribute__((vector_size(16)));

#define AS_GLOBAL __attribute__((address_space(1)))
#define AS_LOCAL  __attribute__((address_space(3)))

#if defined(__has_builtin)
#  if __has_builtin(__builtin_amdgcn_global_load_async_to_lds_b128)
#    define HAVE_ASYNC_LDS 1
#  endif
#  if __has_builtin(__builtin_amdgcn_s_wait_asynccnt)
#    define HAVE_WAIT_ASYNC 1
#  endif
#endif
#ifndef HAVE_ASYNC_LDS
#  define HAVE_ASYNC_LDS 0
#endif
#ifndef HAVE_WAIT_ASYNC
#  define HAVE_WAIT_ASYNC 0
#endif

// WMMA 16x16x32 bf16 fragment K order (contiguous, matches B operand):
// within a lane's 16 values for K-block base k0:
//   slot t in [0,16) -> K = k0 + lhalf*16 + t
__device__ __forceinline__ int frag_koff(int s) {
    int lh = s >> 4;          // lane half this slot belongs to
    int t  = s & 15;
    return lh * 16 + t;
}

// ---------------------------------------------------------------------------
// K0: per-channel batch stats of x -> scale1/shift1  (bn1 folded affine)
// ---------------------------------------------------------------------------
__global__ __launch_bounds__(256)
void bn1_stats_kernel(const float* __restrict__ x,
                      const float* __restrict__ gamma1,
                      const float* __restrict__ beta1,
                      float* __restrict__ scale1,
                      float* __restrict__ shift1)
{
    __shared__ float ssum[256];
    __shared__ float ssq[256];
    const int c   = blockIdx.x;          // 0..511
    const int tid = threadIdx.x;
    float s = 0.f, q = 0.f;
    for (int i = tid; i < NIMG * 256; i += 256) {
        int nn = i >> 8, hw = (i & 255) << 2;
        float4 v = *(const float4*)(x + (((size_t)nn * IN_C + c) << 10) + hw);
        s += v.x + v.y + v.z + v.w;
        q += v.x * v.x + v.y * v.y + v.z * v.z + v.w * v.w;
    }
    ssum[tid] = s; ssq[tid] = q;
    __syncthreads();
    for (int off = 128; off > 0; off >>= 1) {
        if (tid < off) { ssum[tid] += ssum[tid + off]; ssq[tid] += ssq[tid + off]; }
        __syncthreads();
    }
    if (tid == 0) {
        const float inv_n = 1.f / (float)(NIMG * HW);
        float mean = ssum[0] * inv_n;
        float var  = ssq[0] * inv_n - mean * mean;
        float sc   = gamma1[c] * rsqrtf(var + EPSV);
        scale1[c]  = sc;
        shift1[c]  = beta1[c] - mean * sc;
    }
}

// ---------------------------------------------------------------------------
// K1a: pack w1 (128x512 f32) -> bf16, WMMA-A fragment order:
//      w1p[kb(16)][row(128)][32]  (lane reads 16 contiguous bf16 = 32B)
// ---------------------------------------------------------------------------
__global__ __launch_bounds__(256)
void w1_pack_kernel(const float* __restrict__ w1, __bf16* __restrict__ w1p)
{
    int e = blockIdx.x * 256 + threadIdx.x;          // 16*128*32 = 65536
    if (e >= 16 * MID_C * 32) return;
    int s   = e & 31;
    int row = (e >> 5) & 127;
    int kb  = e >> 12;
    w1p[e] = (__bf16)w1[(size_t)row * IN_C + (kb << 5) + frag_koff(s)];
}

// ---------------------------------------------------------------------------
// K1b: pack w2 (OIHW 32x128x3x3 f32) -> w2p[tap(9)][kb(4)][row(32)][32] bf16
//      + precompute bn2 folded affine scale2/shift2 (128 channels)
// ---------------------------------------------------------------------------
__global__ __launch_bounds__(256)
void w2_pack_kernel(const float* __restrict__ w2, __bf16* __restrict__ w2p,
                    const float* __restrict__ g2, const float* __restrict__ b2,
                    const float* __restrict__ m2, const float* __restrict__ v2,
                    float* __restrict__ scale2, float* __restrict__ shift2)
{
    int e = blockIdx.x * 256 + threadIdx.x;          // 9*4*32*32 = 36864
    if (blockIdx.x == 0 && threadIdx.x < MID_C) {
        int p = threadIdx.x;
        float sc = g2[p] * rsqrtf(v2[p] + EPSV);
        scale2[p] = sc;
        shift2[p] = b2[p] - m2[p] * sc;
    }
    if (e >= 9 * 4 * OUT_G * 32) return;
    int s   = e & 31;
    int row = (e >> 5) & 31;
    int kb  = (e >> 10) & 3;
    int tap = e >> 12;
    int ky = tap / 3, kx = tap % 3;
    int c  = (kb << 5) + frag_koff(s);
    w2p[e] = (__bf16)w2[((size_t)row * MID_C + c) * 9 + ky * 3 + kx];
}

// ---------------------------------------------------------------------------
// K2: fused  [copy x -> out[:, :512]]  +  bn1+relu  +  1x1 conv (WMMA bf16)
//     + bn2+relu  -> h2 (bf16, [n][pos][c])
// One workgroup = one image n, one 32-wide hw tile. 8 waves cover M=128;
// each wave keeps two accumulators so every A fragment feeds two WMMAs.
// ---------------------------------------------------------------------------
__global__ __launch_bounds__(256, 2)
void gemm1_kernel(const float* __restrict__ x,
                  const __bf16* __restrict__ w1p,
                  const float* __restrict__ scale1,
                  const float* __restrict__ shift1,
                  const float* __restrict__ scale2,
                  const float* __restrict__ shift2,
                  float* __restrict__ out,
                  __bf16* __restrict__ h2)
{
    __shared__ alignas(64) __bf16 Bs[32 * IN_C];     // [j(32)][c(512)], 32 KB

    const int wg  = blockIdx.x;               // 0..4095
    const int n   = wg >> 5;                  // image
    const int hw0 = (wg & 31) << 5;           // 32-position tile base
    const int tid = threadIdx.x;

    const float* xim   = x   + (size_t)n * IN_C * HW;
    float*       outim = out + (size_t)n * OUT_C * HW;

    // Cooperative float4 load of 512x32 x-tile; doubles as the concat copy.
    for (int e = tid; e < IN_C * 8; e += 256) {
        int c = e >> 3, j4 = (e & 7) << 2;
        float4 v = *(const float4*)(xim + c * HW + hw0 + j4);
        *(float4*)(outim + c * HW + hw0 + j4) = v;         // out[:, :512] = x
        float sc = scale1[c], sh = shift1[c];
        Bs[(j4 + 0) * IN_C + c] = (__bf16)fmaxf(fmaf(v.x, sc, sh), 0.f);
        Bs[(j4 + 1) * IN_C + c] = (__bf16)fmaxf(fmaf(v.y, sc, sh), 0.f);
        Bs[(j4 + 2) * IN_C + c] = (__bf16)fmaxf(fmaf(v.z, sc, sh), 0.f);
        Bs[(j4 + 3) * IN_C + c] = (__bf16)fmaxf(fmaf(v.w, sc, sh), 0.f);
    }
    __syncthreads();

    const int wave  = tid >> 5;
    const int lane  = tid & 31;
    const int m0    = wave << 4;              // 16 p-rows per wave
    const int lhalf = lane >> 4;              // 0 | 1
    const int lmod  = lane & 15;
    const int arow  = m0 + lmod;              // A-matrix row (M = lane % 16)

    v8f acc0 = {}, acc1 = {};
    #pragma unroll 4
    for (int kb = 0; kb < 16; ++kb) {
        v16bf a  = *(const v16bf*)(w1p + (((kb << 7) + arow) << 5) + (lhalf << 4));
        v16bf b0 = *(const v16bf*)(Bs + lmod * IN_C + (kb << 5) + (lhalf << 4));
        v16bf b1 = *(const v16bf*)(Bs + (lmod + 16) * IN_C + (kb << 5) + (lhalf << 4));
        acc0 = __builtin_amdgcn_wmma_f32_16x16x32_bf16(false, a, false, b0,
                                                       (short)0, acc0, false, false);
        acc1 = __builtin_amdgcn_wmma_f32_16x16x32_bf16(false, a, false, b1,
                                                       (short)0, acc1, false, false);
    }

    // bn2 (precomputed affine) + relu -> packed 16B bf16 stores, [pos][c] layout
    v8bf hv0, hv1;
    #pragma unroll
    for (int r = 0; r < 8; ++r) {
        int m = m0 + (lhalf << 3) + r;        // C/D layout: M = r + 8*(lane/16)
        hv0[r] = (__bf16)fmaxf(fmaf(acc0[r], scale2[m], shift2[m]), 0.f);
        hv1[r] = (__bf16)fmaxf(fmaf(acc1[r], scale2[m], shift2[m]), 0.f);
    }
    __bf16* h2p = h2 + (((size_t)n * HW + hw0 + lmod) << 7) + m0 + (lhalf << 3);
    *(v8bf*)(h2p)              = hv0;
    *(v8bf*)(h2p + (16 << 7))  = hv1;
}

// ---------------------------------------------------------------------------
// K3: 3x3 conv (pad 1) as 9 shifted WMMA-GEMMs, K=128.
// One workgroup = one image n, FOUR output rows. 8 waves = 2 M-tiles x 4 rows;
// each wave owns one output row and both 16-wide x-halves (A frags reused 2x).
// LDS halo tile [6 rows][34 x][128 c] bf16 staged via async global->LDS copy.
// ---------------------------------------------------------------------------
__global__ __launch_bounds__(256, 2)
void conv2_kernel(const __bf16* __restrict__ h2,
                  const __bf16* __restrict__ w2p,
                  float* __restrict__ out)
{
    __shared__ alignas(64) __bf16 Bs[6 * 34 * MID_C];   // ~51 KB

    const int wg  = blockIdx.x;               // 0..1023
    const int n   = wg >> 3;
    const int y0  = (wg & 7) << 2;            // output rows y0..y0+3
    const int tid = threadIdx.x;
    const __bf16* h2im = h2 + ((size_t)n * HW << 7);    // [pos][c]

    // Stage rows y0-1..y0+4, x in [-1,32], in 16-byte (8 x bf16) units.
    for (int u = tid; u < 6 * 34 * 16; u += 256) {
        int r   = u / 544;                    // 0..5 -> y = y0-1+r
        int rem = u - r * 544;
        int xr  = rem >> 4;                   // 0..33 -> x = xr-1
        int ck  = (rem & 15) << 3;            // c chunk base
        int xx = xr - 1, yy = y0 - 1 + r;
        int lidx = ((r * 34 + xr) << 7) + ck;
        if (xx >= 0 && xx < 32 && yy >= 0 && yy < 32) {
            const __bf16* src = h2im + (((yy << 5) + xx) << 7) + ck;
#if HAVE_ASYNC_LDS
            __builtin_amdgcn_global_load_async_to_lds_b128(
                (AS_GLOBAL v4i*)src, (AS_LOCAL v4i*)(Bs + lidx), 0, 0);
#else
            *(v8bf*)(Bs + lidx) = *(const v8bf*)src;
#endif
        } else {
            v8bf z = {};
            *(v8bf*)(Bs + lidx) = z;
        }
    }
#if HAVE_ASYNC_LDS
#  if HAVE_WAIT_ASYNC
    __builtin_amdgcn_s_wait_asynccnt(0);
#  else
    asm volatile("s_wait_asynccnt 0" ::: "memory");
#  endif
#endif
    __syncthreads();

    const int wave  = tid >> 5;
    const int lane  = tid & 31;
    const int mt    = wave >> 2;              // 0..1 -> o base 16*mt
    const int nt    = wave & 3;               // 0..3 -> output row y0+nt
    const int oy    = y0 + nt;
    const int lhalf = lane >> 4;
    const int lmod  = lane & 15;
    const int orow  = (mt << 4) + lmod;

    v8f acc0 = {}, acc1 = {};                 // x-halves [0,16) and [16,32)
    #pragma unroll
    for (int tap = 0; tap < 9; ++tap) {
        const int ky = tap / 3, kx = tap % 3;
        const int r  = nt + ky;               // LDS row index 0..5
        #pragma unroll
        for (int kb = 0; kb < 4; ++kb) {
            v16bf a  = *(const v16bf*)(w2p + (((tap * 4 + kb) * OUT_G + orow) << 5)
                                           + (lhalf << 4));
            v16bf b0 = *(const v16bf*)(Bs + (((r * 34) + (lmod + kx)) << 7)
                                          + (kb << 5) + (lhalf << 4));
            v16bf b1 = *(const v16bf*)(Bs + (((r * 34) + (16 + lmod + kx)) << 7)
                                          + (kb << 5) + (lhalf << 4));
            acc0 = __builtin_amdgcn_wmma_f32_16x16x32_bf16(false, a, false, b0,
                                                           (short)0, acc0, false, false);
            acc1 = __builtin_amdgcn_wmma_f32_16x16x32_bf16(false, a, false, b1,
                                                           (short)0, acc1, false, false);
        }
    }

    float* outim = out + (size_t)n * OUT_C * HW + (size_t)IN_C * HW;
    const int pos = (oy << 5) + lmod;
    #pragma unroll
    for (int r8 = 0; r8 < 8; ++r8) {
        int o = (mt << 4) + r8 + (lhalf << 3);
        outim[o * HW + pos]      = acc0[r8];
        outim[o * HW + pos + 16] = acc1[r8];
    }
}

// ---------------------------------------------------------------------------
extern "C" void kernel_launch(void* const* d_in, const int* in_sizes, int n_in,
                              void* d_out, int out_size, void* d_ws, size_t ws_size,
                              hipStream_t stream)
{
    const float* x      = (const float*)d_in[0];
    const float* gamma1 = (const float*)d_in[1];
    const float* beta1  = (const float*)d_in[2];
    const float* w1     = (const float*)d_in[3];
    const float* gamma2 = (const float*)d_in[4];
    const float* beta2  = (const float*)d_in[5];
    const float* mean2  = (const float*)d_in[6];
    const float* var2   = (const float*)d_in[7];
    const float* w2     = (const float*)d_in[8];
    float* out = (float*)d_out;

    char* ws = (char*)d_ws;
    float*  scale1 = (float*)(ws);                    //  512 f32
    float*  shift1 = (float*)(ws + 2048);             //  512 f32
    float*  scale2 = (float*)(ws + 4096);             //  128 f32
    float*  shift2 = (float*)(ws + 4608);             //  128 f32
    __bf16* w1p    = (__bf16*)(ws + 8192);            //  65536 bf16 = 128 KB
    __bf16* w2p    = (__bf16*)(ws + 8192 + 131072);   //  36864 bf16 = 72 KB
    __bf16* h2     = (__bf16*)(ws + 262144);          //  128*1024*128 bf16 = 32 MB

    bn1_stats_kernel<<<IN_C, 256, 0, stream>>>(x, gamma1, beta1, scale1, shift1);
    w1_pack_kernel<<<(16 * MID_C * 32 + 255) / 256, 256, 0, stream>>>(w1, w1p);
    w2_pack_kernel<<<(9 * 4 * OUT_G * 32 + 255) / 256, 256, 0, stream>>>(
        w2, w2p, gamma2, beta2, mean2, var2, scale2, shift2);
    gemm1_kernel<<<NIMG * (HW / 32), 256, 0, stream>>>(x, w1p, scale1, shift1,
                                                       scale2, shift2, out, h2);
    conv2_kernel<<<NIMG * 8, 256, 0, stream>>>(h2, w2p, out);
}
